// EMAU_11184094839131
// MI455X (gfx1250) — matrix operand order repaired and checked by reading, not compiled, not run on hardware
//
#include <hip/hip_runtime.h>
#include <hip/hip_bf16.h>

typedef _Float16 half_t;
typedef __attribute__((ext_vector_type(2)))  half_t half2t;
typedef __attribute__((ext_vector_type(8)))  half_t half8;
typedef __attribute__((ext_vector_type(16))) half_t v16h;
typedef __attribute__((ext_vector_type(8)))  float  v8f;
typedef __attribute__((ext_vector_type(4)))  unsigned int v4u;
typedef __attribute__((ext_vector_type(8)))  int v8i;
typedef __attribute__((ext_vector_type(4)))  int v4i;

typedef __attribute__((address_space(1))) char l_gchar;   // (unused, kept for casts)
typedef __attribute__((address_space(3))) char l_lchar;
typedef __attribute__((address_space(1))) v4i gv4i;       // global 128-bit chunk
typedef __attribute__((address_space(3))) v4i lv4i;       // LDS 128-bit chunk

#define BB 16      // batch
#define CC 512     // channels
#define NN 4096    // h*w
#define KK 64      // em bases

// GEMM tiling: 128x128 block tile, K-step 32, 8 waves (2x4), 64x32 per wave
#define BM 128
#define BN 128
#define BKT 32
#define LDA (BKT + 8)   // 40 halfs -> 80B row stride (16B aligned)
#define LDBT (BKT + 8)  // Bs stored transposed [BN][BKT+8]

// ---- CDNA5 data-mover feature detection (compile-safe fallbacks) -----------
#if __has_builtin(__builtin_amdgcn_tensor_load_to_lds) && \
    __has_builtin(__builtin_amdgcn_s_wait_tensorcnt)
#define HAVE_TDM 1
#else
#define HAVE_TDM 0
#endif
#if __has_include(<hip/amd_detail/amd_gfx1250_TDM.h>)
#define TDM_6ARG 1
#else
#define TDM_6ARG 0
#endif
#if __has_builtin(__builtin_amdgcn_global_load_async_to_lds_b128) && \
    __has_builtin(__builtin_amdgcn_s_wait_asynccnt)
#define HAVE_ASYNC 1
#else
#define HAVE_ASYNC 0
#endif

#if HAVE_ASYNC
__device__ __forceinline__ void async_b128(const half_t* g, half_t* l) {
  __builtin_amdgcn_global_load_async_to_lds_b128((gv4i*)g, (lv4i*)l, 0, 0);
}
#endif

#if HAVE_TDM
// TDM: DMA a full BM x BKT f16 tile (row stride K elems) into LDS with
// 8-half padding per 32-half row (matches LDA=40).  Issued once per block.
__device__ __forceinline__ void tdm_load_a_tile(const half_t* g, unsigned lds_off, int K) {
  unsigned long long ga = (unsigned long long)(size_t)g;
  v4u g0 = { 1u,                                   // count=1 valid descriptor
             lds_off,                              // LDS byte address
             (unsigned)(ga & 0xffffffffu),
             (unsigned)((ga >> 32) & 0x1ffffffu) | (2u << 30) };  // addr hi | type=2
  v8i g1;
  g1[0] = (1 << 16)     // data_size = 2 bytes
        | (1 << 20)     // pad_enable
        | (3 << 22)     // pad_interval: 16 DWORDs (= one 32-half row)
        | (3 << 25);    // pad_amount: 4 DWORDs (= 8 halfs)
  g1[1] = (int)(32u  << 16);   // tensor_dim0 = 32  (bits 79:48, low half)
  g1[2] = (int)(128u << 16);   // tensor_dim0 hi=0 | tensor_dim1 = 128
  g1[3] = (int)(32u  << 16);   // tensor_dim1 hi=0 | tile_dim0 = 32
  g1[4] = 128;                 // tile_dim1 = 128, tile_dim2 = 0
  g1[5] = K;                   // tensor_dim0_stride (elements)
  g1[6] = 0;
  g1[7] = 0;
  v4i gz = {0, 0, 0, 0};
#if TDM_6ARG
  v8i gz8 = {0, 0, 0, 0, 0, 0, 0, 0};
  __builtin_amdgcn_tensor_load_to_lds(g0, g1, gz, gz, gz8, 0);
#else
  __builtin_amdgcn_tensor_load_to_lds(g0, g1, gz, gz, 0);
#endif
}
#endif

// ---------------------------------------------------------------------------
// f16 WMMA GEMM:  out = epilogue( A[MxK] * B[KxN] )
// MODEA: 0 = A row-major [M][K];  1 = A(m,k) = src[k*M + m]
// MODEB: 0 = B row-major [K][N];  1 = B(k,n) = src[n*K + k]
// EPI:   0 = f32 out;  1 = +bias -> f16;  2 = relu -> f16;
//        3 = *scale+shift +resid relu -> f32
// ---------------------------------------------------------------------------
template<int MODEA, int MODEB, int EPI>
__global__ __launch_bounds__(256) void gemm_wmma(
    const half_t* __restrict__ A, const half_t* __restrict__ B,
    int M, int N, int K,
    const float* __restrict__ bias,
    const float* __restrict__ scale, const float* __restrict__ shift,
    const float* __restrict__ resid,
    float* __restrict__ outF, half_t* __restrict__ outH,
    long long sA, long long sB, long long sR, long long sF, long long sH)
{
  __shared__ __align__(16) half_t As[BM * LDA];
  __shared__ __align__(16) half_t Bs[BN * LDBT];   // transposed: Bs[n][k]

  const int b = blockIdx.z;
  A += (size_t)b * sA;
  B += (size_t)b * sB;
  if (resid) resid += (size_t)b * sR;
  if (outF)  outF  += (size_t)b * sF;
  if (outH)  outH  += (size_t)b * sH;

  const int t    = threadIdx.x;
  const int lane = t & 31;
  const int wave = t >> 5;
  const int wm   = wave & 1;       // 2 waves along M
  const int wn   = wave >> 1;      // 4 waves along N
  const int m0   = blockIdx.y * BM;
  const int n0   = blockIdx.x * BN;
  const int l16  = lane & 15;
  const int grp  = lane >> 4;

  v8f acc[4][2] = {};

#if HAVE_TDM
  const bool tdm_ok = (MODEA == 0) && ((M & (BM - 1)) == 0);
#endif

  for (int kb = 0; kb < K; kb += BKT) {
    bool used_async = false;
    // ---- stage A tile into LDS (row-major [BM][LDA]) ----
    if (MODEA == 0) {
#if HAVE_TDM
      if (tdm_ok) {
        if (wave == 0) {
          tdm_load_a_tile(A + (size_t)m0 * K + kb,
                          (unsigned)(unsigned long long)(l_lchar*)As, K);
          __builtin_amdgcn_s_wait_tensorcnt(0);
        }
      } else
#endif
      {
        for (int it = 0; it < 2; it++) {
          int c   = t + (it << 8);            // 512 chunks of 8 halfs
          int row = c >> 2;
          int kc  = (c & 3) << 3;
          int gm  = m0 + row; if (gm >= M) gm = M - 1;
#if HAVE_ASYNC
          async_b128(A + (size_t)gm * K + (kb + kc), &As[row * LDA + kc]);
          used_async = true;
#else
          *(half8*)(&As[row * LDA + kc]) = *(const half8*)(A + (size_t)gm * K + (kb + kc));
#endif
        }
      }
    } else {
      for (int i = 0; i < 16; i++) {
        int idx = t + (i << 8);
        int row = idx & 127;                  // consecutive t -> consecutive m (coalesced)
        int kk  = idx >> 7;
        int gm  = m0 + row; if (gm >= M) gm = M - 1;
        As[row * LDA + kk] = A[(size_t)(kb + kk) * M + gm];
      }
    }
    // ---- stage B tile into LDS, transposed: Bs[n][k] ----
    if (MODEB == 0) {
      for (int it = 0; it < 2; it++) {
        int c  = t + (it << 8);               // 512 chunks (32 k-rows x 16 n-chunks)
        int kk = c >> 4;
        int nc = (c & 15) << 3;
        int gn = n0 + nc;
        half8 v = {};
        if (gn < N) v = *(const half8*)(B + (size_t)(kb + kk) * N + gn);
        for (int j = 0; j < 8; j++) Bs[(nc + j) * LDBT + kk] = v[j];
      }
      if (kb + BKT < K) {
        int gn = n0 + (t & 127); if (gn >= N) gn = N - 1;
        __builtin_prefetch((const void*)(B + (size_t)(kb + BKT) * N + gn), 0, 1);
      }
    } else {
      for (int it = 0; it < 2; it++) {
        int c  = t + (it << 8);               // 512 chunks (128 n-rows x 4 k-chunks)
        int nn = c >> 2;
        int kc = (c & 3) << 3;
        int gn = n0 + nn; if (gn >= N) gn = N - 1;
#if HAVE_ASYNC
        async_b128(B + (size_t)gn * K + (kb + kc), &Bs[nn * LDBT + kc]);
        used_async = true;
#else
        *(half8*)(&Bs[nn * LDBT + kc]) = *(const half8*)(B + (size_t)gn * K + (kb + kc));
#endif
      }
    }
#if HAVE_ASYNC
    if (used_async) __builtin_amdgcn_s_wait_asynccnt(0);
#else
    (void)used_async;
#endif
    __syncthreads();

    // ---- A fragments: lanes 0-15: K{0-7,16-23}, lanes 16-31: K{8-15,24-31}
    v16h af[4];
    for (int mi = 0; mi < 4; mi++) {
      int mr = wm * 64 + mi * 16 + l16;
      half8 lo = *(const half8*)(&As[mr * LDA + grp * 8]);
      half8 hi = *(const half8*)(&As[mr * LDA + 16 + grp * 8]);
      for (int i = 0; i < 8; i++) { af[mi][i] = lo[i]; af[mi][8 + i] = hi[i]; }
    }
    // ---- B fragments from transposed Bs: lane = N, lanes 0-15: K{0-7,16-23}, lanes 16-31: K{8-15,24-31}
    v16h bf[2];
    for (int ni = 0; ni < 2; ni++) {
      int nc = wn * 32 + ni * 16 + l16;
      half8 lo = *(const half8*)(&Bs[nc * LDBT + grp * 8]);
      half8 hi = *(const half8*)(&Bs[nc * LDBT + 16 + grp * 8]);
      for (int i = 0; i < 8; i++) { bf[ni][i] = lo[i]; bf[ni][8 + i] = hi[i]; }
    }
    for (int mi = 0; mi < 4; mi++)
      for (int ni = 0; ni < 2; ni++)
        acc[mi][ni] = __builtin_amdgcn_wmma_f32_16x16x32_f16(
            false, af[mi], false, bf[ni], (short)0, acc[mi][ni], false, false);
    __syncthreads();
  }

  // ---- epilogue: C/D layout: VGPR j -> M = j + 8*grp, N = lane%16
  for (int mi = 0; mi < 4; mi++)
    for (int ni = 0; ni < 2; ni++)
      for (int j = 0; j < 8; j++) {
        int row = m0 + wm * 64 + mi * 16 + j + grp * 8;
        int col = n0 + wn * 32 + ni * 16 + l16;
        if (row < M && col < N) {
          float v = acc[mi][ni][j];
          if constexpr (EPI == 1) v += bias[row];
          if constexpr (EPI == 3) v = v * scale[row] + shift[row] + resid[(size_t)row * N + col];
          if constexpr (EPI == 2 || EPI == 3) v = fmaxf(v, 0.0f);
          size_t o = (size_t)row * N + col;
          if constexpr (EPI == 0 || EPI == 3) outF[o] = v;
          else                                outH[o] = (half_t)v;
        }
      }
}

// ---------------------------------------------------------------------------
// Row softmax over k (64) + f16 z output + column-sum (over n) accumulation
// ---------------------------------------------------------------------------
__global__ __launch_bounds__(256) void softmax_kernel(
    const float* __restrict__ logits, half_t* __restrict__ z, float* __restrict__ zsum)
{
  __shared__ float ssum[KK];
  const int b    = blockIdx.z;
  const int n    = blockIdx.x * 8 + (threadIdx.x >> 5);
  const int lane = threadIdx.x & 31;
  if (threadIdx.x < KK) ssum[threadIdx.x] = 0.0f;
  __syncthreads();

  const float2 v = ((const float2*)(logits + ((size_t)b * NN + n) * KK))[lane];
  float m = fmaxf(v.x, v.y);
  for (int off = 16; off > 0; off >>= 1) m = fmaxf(m, __shfl_xor(m, off, 32));
  float e0 = expf(v.x - m), e1 = expf(v.y - m);
  float s = e0 + e1;
  for (int off = 16; off > 0; off >>= 1) s += __shfl_xor(s, off, 32);
  float z0 = e0 / s, z1 = e1 / s;

  half2t h; h[0] = (half_t)z0; h[1] = (half_t)z1;
  ((half2t*)(z + ((size_t)b * NN + n) * KK))[lane] = h;

  atomicAdd(&ssum[2 * lane],     z0);
  atomicAdd(&ssum[2 * lane + 1], z1);
  __syncthreads();
  if (threadIdx.x < KK) atomicAdd(&zsum[b * KK + threadIdx.x], ssum[threadIdx.x]);
}

// ---------------------------------------------------------------------------
// mu column update: divide by (eps + colsum(z)), l2-normalize over c, emit f16
// ---------------------------------------------------------------------------
__global__ __launch_bounds__(256) void mu_norm_kernel(
    const float* __restrict__ mu_raw, const float* __restrict__ zsum,
    half_t* __restrict__ mu_h)
{
  __shared__ float red[256];
  const int b = blockIdx.z, k = blockIdx.x, t = threadIdx.x;
  const float zden = 1.0f / (1e-6f + zsum[b * KK + k]);
  size_t base = ((size_t)b * CC) * KK + k;
  float v0 = mu_raw[base + (size_t)t * KK] * zden;
  float v1 = mu_raw[base + (size_t)(t + 256) * KK] * zden;
  red[t] = v0 * v0 + v1 * v1;
  __syncthreads();
  for (int s = 128; s > 0; s >>= 1) { if (t < s) red[t] += red[t + s]; __syncthreads(); }
  float inv = 1.0f / (1e-6f + sqrtf(red[0]));
  mu_h[base + (size_t)t * KK]         = (half_t)(v0 * inv);
  mu_h[base + (size_t)(t + 256) * KK] = (half_t)(v1 * inv);
}

// ---------------------------------------------------------------------------
// prep kernels
// ---------------------------------------------------------------------------
__global__ void prep_weights(const float* __restrict__ w1, const float* __restrict__ w2,
                             const float* __restrict__ gamma, const float* __restrict__ beta,
                             const float* __restrict__ mean, const float* __restrict__ var,
                             half_t* __restrict__ w1h, half_t* __restrict__ w2h,
                             float* __restrict__ bninv, float* __restrict__ bnsh)
{
  int i = blockIdx.x * blockDim.x + threadIdx.x;
  if (i < CC * CC) { w1h[i] = (half_t)w1[i]; w2h[i] = (half_t)w2[i]; }
  if (i < CC) {
    float inv = gamma[i] * rsqrtf(var[i] + 1e-5f);
    bninv[i] = inv;
    bnsh[i]  = beta[i] - mean[i] * inv;
  }
}

__global__ void mu_init_kernel(const float* __restrict__ mu0, half_t* __restrict__ mu_h)
{
  int i = blockIdx.x * blockDim.x + threadIdx.x;   // B*C*K
  mu_h[i] = (half_t)mu0[i % (CC * KK)];
}

__global__ void conv_x_kernel(const float* __restrict__ x, half_t* __restrict__ xh, size_t n)
{
  size_t i = (size_t)blockIdx.x * blockDim.x + threadIdx.x;
  if (i < n) xh[i] = (half_t)x[i];
}

__global__ void zero_kernel(float* __restrict__ p, int n)
{
  int i = blockIdx.x * blockDim.x + threadIdx.x;
  if (i < n) p[i] = 0.0f;
}

// ---------------------------------------------------------------------------
extern "C" void kernel_launch(void* const* d_in, const int* in_sizes, int n_in,
                              void* d_out, int out_size, void* d_ws, size_t ws_size,
                              hipStream_t stream) {
  (void)in_sizes; (void)n_in; (void)out_size; (void)ws_size;
  const float* x     = (const float*)d_in[0];
  const float* w1    = (const float*)d_in[1];
  const float* b1    = (const float*)d_in[2];
  const float* mu0   = (const float*)d_in[3];
  const float* w2    = (const float*)d_in[4];
  const float* gamma = (const float*)d_in[5];
  const float* beta  = (const float*)d_in[6];
  const float* rmean = (const float*)d_in[7];
  const float* rvar  = (const float*)d_in[8];
  float* out = (float*)d_out;

  char* ws = (char*)d_ws;
  size_t off = 0;
  auto alloc = [&](size_t bytes) -> char* {
    char* p = ws + off;
    off = (off + bytes + 255) & ~(size_t)255;
    return p;
  };
  half_t* x_h    = (half_t*)alloc((size_t)BB * CC * NN * 2);   // reused as xr_h
  half_t* xs_h   = (half_t*)alloc((size_t)BB * CC * NN * 2);
  float*  logits = (float*) alloc((size_t)BB * NN * KK * 4);
  half_t* z_h    = (half_t*)alloc((size_t)BB * NN * KK * 2);
  half_t* mu_h   = (half_t*)alloc((size_t)BB * CC * KK * 2);
  float*  mu_raw = (float*) alloc((size_t)BB * CC * KK * 4);
  float*  zsum   = (float*) alloc((size_t)BB * KK * 4);
  half_t* w1h    = (half_t*)alloc((size_t)CC * CC * 2);
  half_t* w2h    = (half_t*)alloc((size_t)CC * CC * 2);
  float*  bninv  = (float*) alloc(CC * 4);
  float*  bnsh   = (float*) alloc(CC * 4);

  const long long sCN = (long long)CC * NN;
  const long long sNK = (long long)NN * KK;
  const long long sCK = (long long)CC * KK;

  prep_weights<<<dim3((CC * CC + 255) / 256), 256, 0, stream>>>(
      w1, w2, gamma, beta, rmean, rvar, w1h, w2h, bninv, bnsh);
  mu_init_kernel<<<dim3((BB * CC * KK) / 256), 256, 0, stream>>>(mu0, mu_h);
  conv_x_kernel<<<dim3((unsigned)(((size_t)BB * CC * NN + 255) / 256)), 256, 0, stream>>>(
      x, x_h, (size_t)BB * CC * NN);

  // conv1: x1 = w1 @ x + b1  -> f16 xs  (M=C, N=n, K=C)
  gemm_wmma<0, 0, 1><<<dim3(NN / BN, CC / BM, BB), 256, 0, stream>>>(
      w1h, x_h, CC, NN, CC, b1, nullptr, nullptr, nullptr,
      nullptr, xs_h, 0, sCN, 0, 0, sCN);

  for (int s = 0; s < 3; s++) {
    zero_kernel<<<dim3((BB * KK + 255) / 256), 256, 0, stream>>>(zsum, BB * KK);
    // logits[n,k] = xs^T @ mu  (M=n, N=k, K=C; A transposed source)
    gemm_wmma<1, 0, 0><<<dim3((KK + BN - 1) / BN, NN / BM, BB), 256, 0, stream>>>(
        xs_h, mu_h, NN, KK, CC, nullptr, nullptr, nullptr, nullptr,
        logits, nullptr, sCN, sCK, 0, sNK, 0);
    softmax_kernel<<<dim3(NN / 8, 1, BB), 256, 0, stream>>>(logits, z_h, zsum);
    // mu_raw[c,k] = xs @ z  (M=C, N=k, K=n)
    gemm_wmma<0, 0, 0><<<dim3((KK + BN - 1) / BN, CC / BM, BB), 256, 0, stream>>>(
        xs_h, z_h, CC, KK, NN, nullptr, nullptr, nullptr, nullptr,
        mu_raw, nullptr, sCN, sNK, 0, sCK, 0);
    mu_norm_kernel<<<dim3(KK, 1, BB), 256, 0, stream>>>(mu_raw, zsum, mu_h);
  }

  // xr = relu(mu @ z^T)  (M=C, N=n, K=k; B transposed source) -> reuse x_h buffer
  gemm_wmma<0, 1, 2><<<dim3(NN / BN, CC / BM, BB), 256, 0, stream>>>(
      mu_h, z_h, CC, NN, KK, nullptr, nullptr, nullptr, nullptr,
      nullptr, x_h, sCK, sNK, 0, 0, sCN);

  // conv2 + BN + residual + relu -> fp32 out
  gemm_wmma<0, 0, 3><<<dim3(NN / BN, CC / BM, BB), 256, 0, stream>>>(
      w2h, x_h, CC, NN, CC, nullptr, bninv, bnsh, x,
      out, nullptr, 0, sCN, sCN, sCN, 0);
}
